// HybridSamplerKernel_65481071395874
// MI455X (gfx1250) — matrix operand order repaired and checked by reading, not compiled, not run on hardware
//
#include <hip/hip_runtime.h>
#include <cstdint>
#include <cstddef>

typedef __bf16 bf16;
typedef __attribute__((ext_vector_type(16))) __bf16 v16bf;
typedef __attribute__((ext_vector_type(8)))  __bf16 v8bf;
typedef __attribute__((ext_vector_type(8)))  float  v8f;
typedef int v4i __attribute__((vector_size(16)));   // matches builtin's expected pointee

#define B_ROWS 4096
#define R_ROWS 4096
#define D_DIM  1024
#define H_DIM  4096
#define GAMMA_F 1.0f

#define AS1 __attribute__((address_space(1)))
#define AS3 __attribute__((address_space(3)))

#if __has_builtin(__builtin_amdgcn_global_load_async_to_lds_b128) && \
    __has_builtin(__builtin_amdgcn_s_wait_asynccnt)
#define HAVE_ASYNC_LDS 1
#else
#define HAVE_ASYNC_LDS 0
#endif

#if HAVE_ASYNC_LDS
#define ASYNC_WAIT(n) __builtin_amdgcn_s_wait_asynccnt(n)
#else
#define ASYNC_WAIT(n) ((void)0)
#endif

constexpr int BM = 128, BN = 128, BK = 32, LDSK = 40;  // LDSK*2B = 80B rows, 16B-aligned

union Frag { v16bf v; v8bf h[2]; };

// ---------------- Kernel 1: f32 -> bf16 cast + row squared-norm ----------------
__global__ __launch_bounds__(256) void cast_rowsq_kernel(const float* __restrict__ src,
                                                         bf16* __restrict__ dst,
                                                         float* __restrict__ rowsq,
                                                         int ncols) {
  int row = blockIdx.x;
  const float* s = src + (size_t)row * ncols;
  bf16* d = dst + (size_t)row * ncols;
  float acc = 0.f;
  for (int c = threadIdx.x; c < ncols; c += 256) {
    float v = s[c];
    acc += v * v;
    d[c] = (bf16)v;
  }
  __shared__ float red[256];
  red[threadIdx.x] = acc;
  __syncthreads();
  for (int off = 128; off > 0; off >>= 1) {
    if ((int)threadIdx.x < off) red[threadIdx.x] += red[threadIdx.x + off];
    __syncthreads();
  }
  if (threadIdx.x == 0) rowsq[row] = red[0];
}

// ---------------- Kernel 2: W1 (R x H, f32) -> W1t (H x R, bf16) ----------------
__global__ __launch_bounds__(256) void transpose_cast_kernel(const float* __restrict__ src,
                                                             bf16* __restrict__ dst,
                                                             int rows, int cols) {
  __shared__ float tile[32][33];
  int bx = blockIdx.x * 32;   // column base in src
  int by = blockIdx.y * 32;   // row base in src
  int tx = threadIdx.x & 31;
  int ty = threadIdx.x >> 5;  // 0..7
  for (int i = ty; i < 32; i += 8)
    tile[i][tx] = src[(size_t)(by + i) * cols + (bx + tx)];
  __syncthreads();
  for (int i = ty; i < 32; i += 8)
    dst[(size_t)(bx + i) * rows + (by + tx)] = (bf16)tile[tx][i];
}

// ---------------- Shared GEMM helpers ----------------
// Stage one 128x32 bf16 tile from global into LDS.
// Async path: global_load_async_to_lds_b128 (ASYNCcnt), 2 insts per thread per tile.
__device__ __forceinline__ void stage_tile(const bf16* __restrict__ g, int ld, int rowBase,
                                           int k0, bf16* sh, int tid, bool pf) {
  #pragma unroll
  for (int c = tid; c < (BM * BK / 8); c += 256) {   // 512 x 16B chunks, 2 per thread
    int row = c >> 2;
    int col = (c & 3) << 3;
    const bf16* gp = &g[(size_t)(rowBase + row) * ld + k0 + col];
    bf16* sp = &sh[row * LDSK + col];
#if HAVE_ASYNC_LDS
    __builtin_amdgcn_global_load_async_to_lds_b128(
        (AS1 v4i*)gp, (AS3 v4i*)sp, 0, 0);
#else
    *(v8bf*)sp = *(const v8bf*)gp;
#endif
    if (pf) __builtin_prefetch(gp + 2 * BK, 0, 0);   // warm L2 two tiles ahead
  }
}

struct Tiles {
  v8f acc[2][4];
  int waveM, waveN, lh, r;
};

__device__ __forceinline__ void mma_step(Tiles& t, const bf16* As, const bf16* Bs) {
  Frag a[2], b[4];
  #pragma unroll
  for (int mt = 0; mt < 2; ++mt) {
    int row = t.waveM * 32 + mt * 16 + t.r;
    a[mt].h[0] = *(const v8bf*)&As[row * LDSK + (t.lh << 3)];        // K: half*8..+8
    a[mt].h[1] = *(const v8bf*)&As[row * LDSK + 16 + (t.lh << 3)];   // K: 16+half*8..+8
  }
  #pragma unroll
  for (int nt = 0; nt < 4; ++nt) {
    int col = t.waveN * 64 + nt * 16 + t.r;
    b[nt].h[0] = *(const v8bf*)&Bs[col * LDSK + (t.lh << 3)];        // K: half*8..+8
    b[nt].h[1] = *(const v8bf*)&Bs[col * LDSK + 16 + (t.lh << 3)];   // K: 16+half*8..+8
  }
  #pragma unroll
  for (int mt = 0; mt < 2; ++mt)
    #pragma unroll
    for (int nt = 0; nt < 4; ++nt)
      t.acc[mt][nt] = __builtin_amdgcn_wmma_f32_16x16x32_bf16(
          false, a[mt].v, false, b[nt].v, (short)0, t.acc[mt][nt], false, false);
}

// Double-buffered K-loop shared by both GEMMs.
// Pipeline: issue async copies for tile t+1, wait for tile t's group (older, in-order),
// barrier, run 8 WMMAs on tile t.
__device__ __forceinline__ void gemm_mainloop(Tiles& t, const bf16* __restrict__ A,
                                              const bf16* __restrict__ Bm,
                                              int blockM, int blockN, int K,
                                              bf16 (*As)[BM * LDSK], bf16 (*Bs)[BN * LDSK],
                                              int tid) {
  int nt = K / BK;
  stage_tile(A,  K, blockM, 0, As[0], tid, true);
  stage_tile(Bm, K, blockN, 0, Bs[0], tid, true);
  for (int tt = 0; tt < nt; ++tt) {
    int cur = tt & 1;
    __syncthreads();                       // all waves done reading buf[1-cur] (tile tt-1)
    if (tt + 1 < nt) {
      bool pf = (tt + 3) < nt;
      stage_tile(A,  K, blockM, (tt + 1) * BK, As[cur ^ 1], tid, pf);
      stage_tile(Bm, K, blockN, (tt + 1) * BK, Bs[cur ^ 1], tid, pf);
      ASYNC_WAIT(4);                       // tile tt's 4 async insts complete (in-order)
    } else {
      ASYNC_WAIT(0);
    }
    __syncthreads();                       // buf[cur] visible to all waves
    mma_step(t, As[cur], Bs[cur]);
  }
}

// ---------------- Kernel 3: GEMM1 + fused RBF epilogue ----------------
// Kout[i,j] = exp(-gamma*(xsq[i]+rsq[j]-2*(x_i . r_j)))  stored as bf16
__global__ __launch_bounds__(256) void gemm_rbf_kernel(const bf16* __restrict__ A,
                                                       const bf16* __restrict__ Bm,
                                                       const float* __restrict__ xsq,
                                                       const float* __restrict__ rsq,
                                                       bf16* __restrict__ Kout,
                                                       int N, int K) {
  __shared__ __align__(16) bf16 As[2][BM * LDSK];
  __shared__ __align__(16) bf16 Bs[2][BN * LDSK];
  int tid = threadIdx.x;
  int wave = tid >> 5, lane = tid & 31;
  Tiles t;
  t.waveM = wave >> 1; t.waveN = wave & 1;
  t.lh = lane >> 4;    t.r = lane & 15;
  #pragma unroll
  for (int mt = 0; mt < 2; ++mt)
    #pragma unroll
    for (int nt = 0; nt < 4; ++nt) t.acc[mt][nt] = (v8f)(0.0f);

  int blockM = blockIdx.y * BM, blockN = blockIdx.x * BN;
  gemm_mainloop(t, A, Bm, blockM, blockN, K, As, Bs, tid);

  #pragma unroll
  for (int mt = 0; mt < 2; ++mt) {
    #pragma unroll
    for (int nt = 0; nt < 4; ++nt) {
      int col = blockN + t.waveN * 64 + nt * 16 + t.r;
      float rs = rsq[col];
      #pragma unroll
      for (int v = 0; v < 8; ++v) {
        int row = blockM + t.waveM * 32 + mt * 16 + v + (t.lh << 3);
        float c = t.acc[mt][nt][v];
        float kv = __expf(-GAMMA_F * (xsq[row] + rs - 2.0f * c));
        Kout[(size_t)row * N + col] = (bf16)kv;
      }
    }
  }
}

// ---------------- Kernel 4: GEMM2 + fused bias+tanh epilogue ----------------
// Hout[i,n] = tanh( (k_i . W1[:,n]) + b1[n] )  stored as bf16; Bm = W1t (H x R)
__global__ __launch_bounds__(256) void gemm_tanh_kernel(const bf16* __restrict__ A,
                                                        const bf16* __restrict__ Bm,
                                                        const float* __restrict__ b1,
                                                        bf16* __restrict__ Hout,
                                                        int N, int K) {
  __shared__ __align__(16) bf16 As[2][BM * LDSK];
  __shared__ __align__(16) bf16 Bs[2][BN * LDSK];
  int tid = threadIdx.x;
  int wave = tid >> 5, lane = tid & 31;
  Tiles t;
  t.waveM = wave >> 1; t.waveN = wave & 1;
  t.lh = lane >> 4;    t.r = lane & 15;
  #pragma unroll
  for (int mt = 0; mt < 2; ++mt)
    #pragma unroll
    for (int nt = 0; nt < 4; ++nt) t.acc[mt][nt] = (v8f)(0.0f);

  int blockM = blockIdx.y * BM, blockN = blockIdx.x * BN;
  gemm_mainloop(t, A, Bm, blockM, blockN, K, As, Bs, tid);

  #pragma unroll
  for (int mt = 0; mt < 2; ++mt) {
    #pragma unroll
    for (int nt = 0; nt < 4; ++nt) {
      int col = blockN + t.waveN * 64 + nt * 16 + t.r;
      float bias = b1[col];
      #pragma unroll
      for (int v = 0; v < 8; ++v) {
        int row = blockM + t.waveM * 32 + mt * 16 + v + (t.lh << 3);
        float hv = tanhf(t.acc[mt][nt][v] + bias);
        Hout[(size_t)row * N + col] = (bf16)hv;
      }
    }
  }
}

// ---------------- Kernel 5: logits (h @ W2 + b2) + softmax ----------------
__global__ __launch_bounds__(256) void logits_softmax_kernel(const bf16* __restrict__ h,
                                                             const float* __restrict__ W2,
                                                             const float* __restrict__ b2,
                                                             float* __restrict__ out) {
  int wave = threadIdx.x >> 5, lane = threadIdx.x & 31;
  int row = blockIdx.x * 8 + wave;
  const bf16* hr = h + (size_t)row * H_DIM;
  float s0 = 0.f, s1 = 0.f;
  for (int c = lane; c < H_DIM; c += 32) {
    float hv = (float)hr[c];
    s0 += hv * W2[c * 2 + 0];
    s1 += hv * W2[c * 2 + 1];
  }
  #pragma unroll
  for (int off = 16; off > 0; off >>= 1) {
    s0 += __shfl_down(s0, off, 32);
    s1 += __shfl_down(s1, off, 32);
  }
  if (lane == 0) {
    float l0 = s0 + b2[0], l1 = s1 + b2[1];
    float m = fmaxf(l0, l1);
    float e0 = __expf(l0 - m), e1 = __expf(l1 - m);
    float inv = 1.0f / (e0 + e1);
    out[row * 2 + 0] = e0 * inv;
    out[row * 2 + 1] = e1 * inv;
  }
}

// ---------------- Host launcher ----------------
extern "C" void kernel_launch(void* const* d_in, const int* in_sizes, int n_in,
                              void* d_out, int out_size, void* d_ws, size_t ws_size,
                              hipStream_t stream) {
  const float* x    = (const float*)d_in[0];
  const float* refs = (const float*)d_in[1];
  const float* W1   = (const float*)d_in[2];
  const float* b1   = (const float*)d_in[3];
  const float* W2   = (const float*)d_in[4];
  const float* b2   = (const float*)d_in[5];
  float* out = (float*)d_out;

  // Workspace carve (bf16 regions first, f32 scalars last; all offsets 16B-aligned)
  uintptr_t base = (uintptr_t)d_ws;
  bf16* xb   = (bf16*)base;                                   // B x D
  bf16* rb   = xb + (size_t)B_ROWS * D_DIM;                   // R x D
  bf16* w1t  = rb + (size_t)R_ROWS * D_DIM;                   // H x R (transposed)
  bf16* kmat = w1t + (size_t)H_DIM * R_ROWS;                  // B x R
  bf16* hbuf = kmat + (size_t)B_ROWS * R_ROWS;                // B x H
  float* xsq = (float*)(hbuf + (size_t)B_ROWS * H_DIM);       // B
  float* rsq = xsq + B_ROWS;                                  // R

  // 1) casts + row norms
  cast_rowsq_kernel<<<B_ROWS, 256, 0, stream>>>(x, xb, xsq, D_DIM);
  cast_rowsq_kernel<<<R_ROWS, 256, 0, stream>>>(refs, rb, rsq, D_DIM);

  // 2) W1 (R x H) -> W1t (H x R) bf16
  transpose_cast_kernel<<<dim3(H_DIM / 32, R_ROWS / 32), 256, 0, stream>>>(W1, w1t, R_ROWS, H_DIM);

  // 3) GEMM1 + RBF epilogue: kmat = exp(-gamma * ||x_i - r_j||^2)
  gemm_rbf_kernel<<<dim3(R_ROWS / BN, B_ROWS / BM), 256, 0, stream>>>(
      xb, rb, xsq, rsq, kmat, R_ROWS, D_DIM);

  // 4) GEMM2 + tanh epilogue: h = tanh(kmat @ W1 + b1)
  gemm_tanh_kernel<<<dim3(H_DIM / BN, B_ROWS / BM), 256, 0, stream>>>(
      kmat, w1t, b1, hbuf, H_DIM, R_ROWS);

  // 5) logits + softmax
  logits_softmax_kernel<<<B_ROWS / 8, 256, 0, stream>>>(hbuf, W2, b2, out);
}
